// EarthAttention3D_70162585748163
// MI455X (gfx1250) — matrix operand order repaired and checked by reading, not compiled, not run on hardware
//
#include <hip/hip_runtime.h>
#include <hip/hip_bf16.h>

typedef __attribute__((ext_vector_type(16))) _Float16 v16h;
typedef __attribute__((ext_vector_type(8)))  float    v8f;
typedef __attribute__((ext_vector_type(4)))  int      v4i;

#define B__   15
#define NW_   64
#define NWIN  (B__*NW_)   // 960
#define NTOK  144
#define DIM   384
#define TDIM  1152
#define HEADS 12
#define HD    32
#define QSCALE 0.17677669529663687f   // 32^-0.5

// ---- WMMA fragment helpers (wave32 layouts per CDNA5 ISA) ----
// A 16x32 f16: lane row = lane&15; halves [k0, k0+7] -> VGPR0-3, [k0+16, k0+23] -> VGPR4-7, k0 = (lane>>4)*8
__device__ __forceinline__ v16h frag2x16(const _Float16* p0, const _Float16* p1) {
  v16h r;
  ((uint4*)&r)[0] = *(const uint4*)p0;
  ((uint4*)&r)[1] = *(const uint4*)p1;
  return r;
}
// B 32x16 f16 via B^T row-major: lane col = lane&15; [k0, k0+7] and [k0+16, k0+23], k0 = (lane>>4)*8
__device__ __forceinline__ v16h frag32c(const _Float16* p) {
  v16h r;
  ((uint4*)&r)[0] = ((const uint4*)p)[0];
  ((uint4*)&r)[1] = ((const uint4*)(p + 16))[0];
  return r;
}
__device__ __forceinline__ v8f wmma16(v16h a, v16h b, v8f c) {
  return __builtin_amdgcn_wmma_f32_16x16x32_f16(false, a, false, b, (short)0, c, false, false);
}
__device__ __forceinline__ v8f zero8() { v8f z = {0.f,0.f,0.f,0.f,0.f,0.f,0.f,0.f}; return z; }

// ---- async global->LDS 16B copy (ASYNCcnt path); safe fallback if builtin absent ----
#if __has_builtin(__builtin_amdgcn_global_load_async_to_lds_b128)
#define HAS_ASYNC_LDS 1
#else
#define HAS_ASYNC_LDS 0
#endif

typedef __attribute__((address_space(1))) v4i* as1_v4i_ptr;
typedef __attribute__((address_space(3))) v4i* as3_v4i_ptr;

__device__ __forceinline__ void async_cp16(const void* g, void* l) {
#if HAS_ASYNC_LDS
  __builtin_amdgcn_global_load_async_to_lds_b128(
      (as1_v4i_ptr)(void*)g, (as3_v4i_ptr)l, 0, 0);
#else
  *(uint4*)l = *(const uint4*)g;
#endif
}
__device__ __forceinline__ void async_wait0() {
#if HAS_ASYNC_LDS
#if __has_builtin(__builtin_amdgcn_s_wait_asynccnt)
  __builtin_amdgcn_s_wait_asynccnt(0);
#else
  asm volatile("s_wait_asynccnt 0x0" ::: "memory");
#endif
#endif
}

// ---- Kernel 0: weights -> f16 transposed (B^T layout, stays L2-resident) ----
__global__ void __launch_bounds__(256)
prep_weights(const float* __restrict__ wqkv, const float* __restrict__ wproj,
             _Float16* __restrict__ wqkvT, _Float16* __restrict__ wprojT) {
  const int T1 = TDIM * DIM;
  const int T2 = DIM * DIM;
  int idx = blockIdx.x * 256 + threadIdx.x;
  if (idx < T1) {
    int n = idx / DIM, k = idx % DIM;
    wqkvT[idx] = (_Float16)wqkv[k * TDIM + n];
  } else if (idx < T1 + T2) {
    int i2 = idx - T1;
    int n = i2 / DIM, k = i2 % DIM;
    wprojT[i2] = (_Float16)wproj[k * DIM + n];
  }
}

// ---- Kernel 1: QKV projection. One block = 48 rows of one window. ----
// Writes q (pre-scaled), k as (bw, h, n, hd) f16;  v transposed as (bw, h, hd, n) f16.
__global__ void __launch_bounds__(256)
qkv_kernel(const float* __restrict__ x, const float* __restrict__ b_qkv,
           const _Float16* __restrict__ wqkvT,
           _Float16* __restrict__ qb, _Float16* __restrict__ kb, _Float16* __restrict__ vtb) {
  __shared__ alignas(16) _Float16 xs[48][392];   // padded: conflict-free A reads
  const int bw = blockIdx.x / 3;
  const int m0 = (blockIdx.x % 3) * 48;
  const int tid = threadIdx.x;

  const float* xg = x + ((size_t)bw * NTOK + m0) * DIM;
  for (int i = tid; i < 48 * DIM; i += 256) {
    xs[i / DIM][i % DIM] = (_Float16)xg[i];
  }
  __syncthreads();

  const int wave = tid >> 5, lane = tid & 31;
  const int lrow = lane & 15, lgrp = lane >> 4;
  const int k0A = lgrp * 8, k0B = lgrp * 8;

  for (int j = 0; j < 9; ++j) {
    const int nt = wave + 8 * j;          // 72 N-tiles across 8 waves
    const int c  = nt * 16 + lrow;        // output column in [0,1152)
    v8f acc[3] = {zero8(), zero8(), zero8()};
    const _Float16* bbase = wqkvT + (size_t)c * DIM + k0B;
    v16h bf = frag32c(bbase);             // double-buffered B stream
    #pragma unroll
    for (int kk = 0; kk < 12; ++kk) {
      v16h bc = bf;
      if (kk < 11) bf = frag32c(bbase + (kk + 1) * 32);   // prefetch next K-step
      #pragma unroll
      for (int mt = 0; mt < 3; ++mt) {
        v16h af = frag2x16(&xs[mt * 16 + lrow][kk * 32 + k0A],
                           &xs[mt * 16 + lrow][kk * 32 + k0A + 16]);
        acc[mt] = wmma16(af, bc, acc[mt]);
      }
    }
    const int which = c / DIM;
    const int cc = c % DIM;
    const int h = cc / HD, hd = cc % HD;
    const float bias = b_qkv[c];
    #pragma unroll
    for (int mt = 0; mt < 3; ++mt) {
      if (which == 2) {
        // V transposed: 8 consecutive m -> one 16B store
        union { _Float16 h[8]; uint4 u; } pk;
        #pragma unroll
        for (int r = 0; r < 8; ++r) pk.h[r] = (_Float16)(acc[mt][r] + bias);
        const int m = m0 + mt * 16 + lgrp * 8;
        *(uint4*)(vtb + (((size_t)bw * HEADS + h) * HD + hd) * NTOK + m) = pk.u;
      } else {
        _Float16* dst = (which == 0 ? qb : kb) +
                        ((size_t)bw * HEADS + h) * NTOK * HD + hd;
        const float s = (which == 0) ? QSCALE : 1.0f;
        #pragma unroll
        for (int r = 0; r < 8; ++r) {
          const int m = m0 + mt * 16 + lgrp * 8 + r;
          dst[(size_t)m * HD] = (_Float16)((acc[mt][r] + bias) * s);
        }
      }
    }
  }
}

// ---- Kernel 2: attention per (window, head). 4 waves, 9 M-strips round-robin. ----
__global__ void __launch_bounds__(128)
attn_kernel(const _Float16* __restrict__ qb, const _Float16* __restrict__ kb,
            const _Float16* __restrict__ vtb,
            const float* __restrict__ bias_table, const int* __restrict__ pos_idx,
            const float* __restrict__ mask, _Float16* __restrict__ ctx) {
  __shared__ alignas(16) _Float16 Qs[NTOK][40];      // 11.5 KB, padded
  __shared__ alignas(16) _Float16 Ks[NTOK][40];      // 11.5 KB
  __shared__ alignas(16) _Float16 Vt[HD][168];       // 10.5 KB, K padded to 160
  __shared__ alignas(16) _Float16 Ps[4][16][168];    // 21.0 KB, per-wave P strip
  const int bwh = blockIdx.x;
  const int bw = bwh / HEADS, h = bwh % HEADS;
  const int b = bw / NW_, w = bw % NW_;
  const int tid = threadIdx.x;

  // ---- stage Q, K (144x32 f16, padded rows of 40) and Vt (32x144, padded 168)
  // via async global->LDS b128 copies (ASYNCcnt), 16B chunks.
  const char* qg = (const char*)(qb + (size_t)bwh * NTOK * HD);
  const char* kg = (const char*)(kb + (size_t)bwh * NTOK * HD);
  const char* vg = (const char*)(vtb + (size_t)bwh * HD * NTOK);
  char* qls = (char*)&Qs[0][0];
  char* kls = (char*)&Ks[0][0];
  char* vls = (char*)&Vt[0][0];
  for (int i = tid; i < 576; i += 128) {   // 144 rows x 4 chunks of 16B
    const int r = i >> 2, p = i & 3;
    async_cp16(qg + i * 16, qls + r * 80 + p * 16);
    async_cp16(kg + i * 16, kls + r * 80 + p * 16);
  }
  for (int i = tid; i < 576; i += 128) {   // 32 rows x 18 chunks of 16B
    const int r = i / 18, p = i % 18;
    async_cp16(vg + i * 16, vls + r * 336 + p * 16);
  }
  for (int i = tid; i < HD * 24; i += 128)               // zero K-pad 144..167
    Vt[i / 24][144 + i % 24] = (_Float16)0.f;
  for (int i = tid; i < 4 * 16 * 24; i += 128)
    Ps[i / (16 * 24)][(i / 24) % 16][144 + i % 24] = (_Float16)0.f;
  async_wait0();
  __syncthreads();

  const int wave = tid >> 5, lane = tid & 31;
  const int lrow = lane & 15, lgrp = lane >> 4;
  const int k0A = lgrp * 8, k0B = lgrp * 8;

  for (int s = wave; s < 9; s += 4) {
    const int mbase = s * 16;
    // --- S = Q K^T : hd=32 => one WMMA per 16x16 tile, B stream double-buffered ---
    v8f sacc[9];
    #pragma unroll
    for (int t = 0; t < 9; ++t) sacc[t] = zero8();
    v16h aq = frag2x16(&Qs[mbase + lrow][k0A], &Qs[mbase + lrow][k0A + 16]);
    v16h bk = frag32c(&Ks[lrow][k0B]);
    #pragma unroll
    for (int t = 0; t < 9; ++t) {
      v16h bc = bk;
      if (t < 8) bk = frag32c(&Ks[(t + 1) * 16 + lrow][k0B]);
      sacc[t] = wmma16(aq, bc, sacc[t]);
    }
    // --- + earth-position bias + mask, softmax over 144 cols ---
    float rmax[8], rsum[8];
    #pragma unroll
    for (int r = 0; r < 8; ++r) { rmax[r] = -1e30f; rsum[r] = 0.f; }
    #pragma unroll
    for (int t = 0; t < 9; ++t) {
      const int n = t * 16 + lrow;
      #pragma unroll
      for (int r = 0; r < 8; ++r) {
        const int m = mbase + lgrp * 8 + r;
        const int pi = pos_idx[m * NTOK + n];
        float v = sacc[t][r] + bias_table[((size_t)pi * NW_ + w) * HEADS + h]
                             + mask[((size_t)b * NTOK + m) * NTOK + n];
        sacc[t][r] = v;
        rmax[r] = fmaxf(rmax[r], v);
      }
    }
    #pragma unroll
    for (int r = 0; r < 8; ++r) {   // row max across the 16 lanes owning the row
      float m = rmax[r];
      m = fmaxf(m, __shfl_xor(m, 1, 16));
      m = fmaxf(m, __shfl_xor(m, 2, 16));
      m = fmaxf(m, __shfl_xor(m, 4, 16));
      m = fmaxf(m, __shfl_xor(m, 8, 16));
      rmax[r] = m;
    }
    #pragma unroll
    for (int t = 0; t < 9; ++t)
      #pragma unroll
      for (int r = 0; r < 8; ++r) {
        float e = __expf(sacc[t][r] - rmax[r]);
        sacc[t][r] = e;
        rsum[r] += e;
      }
    #pragma unroll
    for (int r = 0; r < 8; ++r) {
      float s2 = rsum[r];
      s2 += __shfl_xor(s2, 1, 16);
      s2 += __shfl_xor(s2, 2, 16);
      s2 += __shfl_xor(s2, 4, 16);
      s2 += __shfl_xor(s2, 8, 16);
      rsum[r] = 1.0f / s2;
    }
    // --- reshape P (C layout) -> A layout via per-wave LDS slice ---
    #pragma unroll
    for (int t = 0; t < 9; ++t)
      #pragma unroll
      for (int r = 0; r < 8; ++r)
        Ps[wave][lgrp * 8 + r][t * 16 + lrow] = (_Float16)(sacc[t][r] * rsum[r]);
    asm volatile("s_wait_dscnt 0" ::: "memory");   // same-wave LDS RAW
    // --- O = P V : K padded 144->160, 5 K-steps x 2 hd-tiles, pipelined ---
    v8f cacc[2] = {zero8(), zero8()};
    v16h ap = frag2x16(&Ps[wave][lrow][k0A], &Ps[wave][lrow][k0A + 16]);
    v16h bv0 = frag32c(&Vt[lrow][k0B]);
    v16h bv1 = frag32c(&Vt[16 + lrow][k0B]);
    #pragma unroll
    for (int kk = 0; kk < 5; ++kk) {
      v16h ac = ap, b0 = bv0, b1 = bv1;
      if (kk < 4) {
        const int ko = (kk + 1) * 32;
        ap  = frag2x16(&Ps[wave][lrow][ko + k0A], &Ps[wave][lrow][ko + k0A + 16]);
        bv0 = frag32c(&Vt[lrow][ko + k0B]);
        bv1 = frag32c(&Vt[16 + lrow][ko + k0B]);
      }
      cacc[0] = wmma16(ac, b0, cacc[0]);
      cacc[1] = wmma16(ac, b1, cacc[1]);
    }
    #pragma unroll
    for (int ti = 0; ti < 2; ++ti) {
      const int hd = ti * 16 + lrow;
      #pragma unroll
      for (int r = 0; r < 8; ++r) {
        const int m = mbase + lgrp * 8 + r;
        ctx[((size_t)bw * NTOK + m) * DIM + h * HD + hd] = (_Float16)cacc[ti][r];
      }
    }
  }
}

// ---- Kernel 3: output projection, one block per window, pipelined A/B streams ----
__global__ void __launch_bounds__(256)
proj_kernel(const _Float16* __restrict__ ctx, const _Float16* __restrict__ wprojT,
            const float* __restrict__ b_proj, float* __restrict__ out) {
  const int bw = blockIdx.x;
  const int tid = threadIdx.x;
  const int wave = tid >> 5, lane = tid & 31;
  const int lrow = lane & 15, lgrp = lane >> 4;
  const int k0A = lgrp * 8, k0B = lgrp * 8;
  const _Float16* cg = ctx + (size_t)bw * NTOK * DIM;

  const _Float16* wr0 = wprojT + (size_t)(wave * 48 + 0 * 16 + lrow) * DIM + k0B;
  const _Float16* wr1 = wprojT + (size_t)(wave * 48 + 1 * 16 + lrow) * DIM + k0B;
  const _Float16* wr2 = wprojT + (size_t)(wave * 48 + 2 * 16 + lrow) * DIM + k0B;

  for (int mt = 0; mt < 9; ++mt) {
    v8f acc[3] = {zero8(), zero8(), zero8()};
    const _Float16* arow = cg + (size_t)(mt * 16 + lrow) * DIM + k0A;
    v16h af  = frag2x16(arow, arow + 16);
    v16h bf0 = frag32c(wr0);
    v16h bf1 = frag32c(wr1);
    v16h bf2 = frag32c(wr2);
    #pragma unroll
    for (int kk = 0; kk < 12; ++kk) {
      v16h ac = af, b0 = bf0, b1 = bf1, b2 = bf2;
      if (kk < 11) {
        const int ko = (kk + 1) * 32;
        af  = frag2x16(arow + ko, arow + ko + 16);
        bf0 = frag32c(wr0 + ko);
        bf1 = frag32c(wr1 + ko);
        bf2 = frag32c(wr2 + ko);
      }
      acc[0] = wmma16(ac, b0, acc[0]);
      acc[1] = wmma16(ac, b1, acc[1]);
      acc[2] = wmma16(ac, b2, acc[2]);
    }
    #pragma unroll
    for (int j = 0; j < 3; ++j) {
      const int n = wave * 48 + j * 16 + lrow;
      const float bias = b_proj[n];
      #pragma unroll
      for (int r = 0; r < 8; ++r) {
        const int m = mt * 16 + lgrp * 8 + r;
        out[((size_t)bw * NTOK + m) * DIM + n] = acc[j][r] + bias;
      }
    }
  }
}

extern "C" void kernel_launch(void* const* d_in, const int* in_sizes, int n_in,
                              void* d_out, int out_size, void* d_ws, size_t ws_size,
                              hipStream_t stream) {
  const float* x      = (const float*)d_in[0];
  const float* mask   = (const float*)d_in[1];
  const float* w_qkv  = (const float*)d_in[2];
  const float* b_qkv  = (const float*)d_in[3];
  const float* w_proj = (const float*)d_in[4];
  const float* b_proj = (const float*)d_in[5];
  const float* btab   = (const float*)d_in[6];
  const int*   pidx   = (const int*)d_in[7];
  float* out = (float*)d_out;

  char* p = (char*)d_ws;
  auto carve = [&](size_t bytes) { char* r = p; p += (bytes + 255) & ~(size_t)255; return r; };
  _Float16* wqkvT  = (_Float16*)carve((size_t)TDIM * DIM * 2);
  _Float16* wprojT = (_Float16*)carve((size_t)DIM * DIM * 2);
  const size_t qkv_e = (size_t)NWIN * HEADS * NTOK * HD;  // 53,084,160
  _Float16* qb  = (_Float16*)carve(qkv_e * 2);
  _Float16* kb  = (_Float16*)carve(qkv_e * 2);
  _Float16* vtb = (_Float16*)carve(qkv_e * 2);
  _Float16* ctx = (_Float16*)carve((size_t)NWIN * NTOK * DIM * 2);
  (void)in_sizes; (void)n_in; (void)out_size; (void)ws_size;

  prep_weights<<<(TDIM * DIM + DIM * DIM + 255) / 256, 256, 0, stream>>>(w_qkv, w_proj, wqkvT, wprojT);
  qkv_kernel<<<NWIN * 3, 256, 0, stream>>>(x, b_qkv, wqkvT, qb, kb, vtb);
  attn_kernel<<<NWIN * HEADS, 128, 0, stream>>>(qb, kb, vtb, btab, pidx, mask, ctx);
  proj_kernel<<<NWIN, 256, 0, stream>>>(ctx, wprojT, b_proj, out);
}
